// MPNNet_Atom_51960514347051
// MI455X (gfx1250) — compile-verified, matrix operand order checked by reading, not run on hardware
//
#include <hip/hip_runtime.h>
#include <hip/hip_bf16.h>

// ---------------------------------------------------------------------------
// MPNNet forward for MI455X (gfx1250, wave32, WMMA).
//  - edge-net GEMM [E,128]x[128,4096] via v_wmma_f32_16x16x32_f16,
//    register-blocked 1x4 column tiles per wave (A-fragment reuse x4)
//  - We materialized in f16 (164 MB -> fits 192 MB L2, re-read 4x cheaply)
//  - conv_root / GRU matmuls via WMMA, f16 A/B, f32 accum
//  - scatter-mean via global_atomic_add_f32
// ---------------------------------------------------------------------------

typedef __attribute__((ext_vector_type(16))) _Float16 v16h;
typedef __attribute__((ext_vector_type(8)))  _Float16 v8h;
typedef __attribute__((ext_vector_type(8)))  float    v8f;

#define DIM 64
#define NF  14

__device__ __forceinline__ float lrelu(float v) { return v > 0.f ? v : 0.01f * v; }
__device__ __forceinline__ float sigm(float x)  { return 1.f / (1.f + expf(-x)); }

// Load one 16x32 f16 fragment (A row-fragment or B column-fragment; both use the
// same per-lane pattern). p0 points at (tile row/col 0, k0). ld = K-stride in halfs.
// ISA layout: lanes 0-15 -> M=lane, K chunks {0..7, 16..23}; lanes 16-31 -> K {8..15, 24..31}.
__device__ __forceinline__ v16h load_frag(const _Float16* __restrict__ p0, int ld, int lane) {
    int r  = lane & 15;
    int kb = (lane < 16) ? 0 : 8;
    const _Float16* p = p0 + r * ld + kb;
    union { v16h v; v8h h[2]; } u;
    u.h[0] = *(const v8h*)(p);        // 16B load -> global_load_b128
    u.h[1] = *(const v8h*)(p + 16);
    return u.v;
}

__device__ __forceinline__ v8f wmma16(v16h a, v16h b, v8f c) {
    return __builtin_amdgcn_wmma_f32_16x16x32_f16(false, a, false, b, (short)0, c, false, false);
}

// ---------------------------- utility kernels ------------------------------

__global__ void k_zero(float* __restrict__ p, int n) {
    int i = blockIdx.x * blockDim.x + threadIdx.x;
    if (i < n) p[i] = 0.f;
}

__global__ void k_cast16(_Float16* __restrict__ d, const float* __restrict__ s, int n) {
    int i = blockIdx.x * blockDim.x + threadIdx.x;
    if (i < n) d[i] = (_Float16)s[i];
}

// dst[j*K + k] = src[k*J + j]  (build K-contiguous Bt layout)
__global__ void k_transpose_cast16(_Float16* __restrict__ d, const float* __restrict__ s,
                                   int J, int K) {
    int i = blockIdx.x * blockDim.x + threadIdx.x;
    if (i >= J * K) return;
    int j = i / K, k = i % K;
    d[i] = (_Float16)s[k * J + j];
}

__global__ void k_count(const int* __restrict__ dst, float* __restrict__ cnt, int E) {
    int e = blockIdx.x * blockDim.x + threadIdx.x;
    if (e < E) atomicAdd(&cnt[dst[e]], 1.0f);
}

// ------------------------------ stage kernels ------------------------------

// out = leaky_relu(x @ lin0_w + b)  -> h (f32) and out16 (f16)
__global__ void k_lin0(const float* __restrict__ x, const float* __restrict__ w,
                       const float* __restrict__ b, float* __restrict__ h,
                       _Float16* __restrict__ out16, int N) {
    int idx = blockIdx.x * blockDim.x + threadIdx.x;
    if (idx >= N * DIM) return;
    int n = idx >> 6, d = idx & 63;
    float acc = b[d];
    const float* xp = x + n * NF;
#pragma unroll
    for (int i = 0; i < NF; ++i) acc += xp[i] * w[i * DIM + d];
    acc = lrelu(acc);
    h[idx] = acc;
    out16[idx] = (_Float16)acc;
}

// g = leaky_relu(edge_attr @ enet_w1 + b1) -> f16 [E,128]
__global__ void k_enet1(const float* __restrict__ ea, const float* __restrict__ w1,
                        const float* __restrict__ b1, _Float16* __restrict__ g, int E) {
    int idx = blockIdx.x * blockDim.x + threadIdx.x;
    if (idx >= E * 128) return;
    int e = idx >> 7, k = idx & 127;
    float acc = b1[k];
    const float* ep = ea + e * 4;
#pragma unroll
    for (int i = 0; i < 4; ++i) acc += ep[i] * w1[i * 128 + k];
    g[idx] = (_Float16)lrelu(acc);
}

// We = g @ enet_w2 + b2 -> f16 [E, 4096].
// Register-blocked: one wave computes a 16x64 slab (4 column tiles sharing the
// A fragment), K=128 -> 16 WMMA per wave, A reused 4x per k-step.
__global__ void k_enet2(const _Float16* __restrict__ g, const _Float16* __restrict__ w2t,
                        const float* __restrict__ b2, _Float16* __restrict__ We, int E) {
    int lane = threadIdx.x & 31;
    int tile = blockIdx.x * 8 + (threadIdx.x >> 5);
    int mtiles = E >> 4;
    if (tile >= mtiles * 64) return;            // wave-uniform guard (EXEC stays all-1)
    int mt = tile >> 6;                          // row tile
    int ng = tile & 63;                          // group of 4 column tiles
    const _Float16* A  = g   + (size_t)(mt * 16) * 128;
    const _Float16* B0 = w2t + (size_t)(ng * 64) * 128;
    v8f c0 = {}, c1 = {}, c2 = {}, c3 = {};
#pragma unroll
    for (int kk = 0; kk < 128; kk += 32) {
        v16h a = load_frag(A + kk, 128, lane);
        c0 = wmma16(a, load_frag(B0 + kk,            128, lane), c0);
        c1 = wmma16(a, load_frag(B0 + 16 * 128 + kk, 128, lane), c1);
        c2 = wmma16(a, load_frag(B0 + 32 * 128 + kk, 128, lane), c2);
        c3 = wmma16(a, load_frag(B0 + 48 * 128 + kk, 128, lane), c3);
    }
    int rbase = mt * 16 + ((lane < 16) ? 0 : 8);
    int coll  = lane & 15;
    auto store_tile = [&](v8f c, int nt) {
        int col  = nt * 16 + coll;
        float bv = b2[col];
#pragma unroll
        for (int v = 0; v < 8; ++v)
            We[(size_t)(rbase + v) * 4096 + col] = (_Float16)(c[v] + bv);
    };
    store_tile(c0, ng * 4 + 0);
    store_tile(c1, ng * 4 + 1);
    store_tile(c2, ng * 4 + 2);
    store_tile(c3, ng * 4 + 3);
}

// msg[e,:] = h[src[e]] @ We[e]   and scatter-add into agg[dst[e],:]
// 8 threads per edge, each owns 8 consecutive outputs via v8h (128-bit) loads.
__global__ void k_msg(const float* __restrict__ h, const _Float16* __restrict__ We,
                      const int* __restrict__ src, const int* __restrict__ dst,
                      float* __restrict__ agg, int E) {
    int t = blockIdx.x * blockDim.x + threadIdx.x;
    int e = t >> 3;
    if (e >= E) return;
    int ob = (t & 7) * 8;
    const _Float16* Wp = We + (size_t)e * 4096 + ob;
    const float*    hp = h + (size_t)src[e] * DIM;
    float acc[8] = {0.f, 0.f, 0.f, 0.f, 0.f, 0.f, 0.f, 0.f};
#pragma unroll 4
    for (int i = 0; i < DIM; ++i) {
        float hv = hp[i];                       // broadcast within edge group
        v8h w = *(const v8h*)(Wp + i * 64);     // coalesced 128-bit, L2-resident
#pragma unroll
        for (int k = 0; k < 8; ++k) acc[k] += hv * (float)w[k];
    }
    float* ap = agg + (size_t)dst[e] * DIM + ob;
#pragma unroll
    for (int k = 0; k < 8; ++k) atomicAdd(ap + k, acc[k]);
}

// m = leaky_relu(agg/denom + out@conv_root + bias) -> m16 (f16).  WMMA, K=64.
__global__ void k_conv_m(const _Float16* __restrict__ out16, const _Float16* __restrict__ crt,
                         const float* __restrict__ agg, const float* __restrict__ cnt,
                         const float* __restrict__ cbias, _Float16* __restrict__ m16, int N) {
    int lane = threadIdx.x & 31;
    int tile = blockIdx.x * 8 + (threadIdx.x >> 5);
    int mtiles = N >> 4;
    if (tile >= mtiles * 4) return;
    int mt = tile >> 2, nt = tile & 3;
    const _Float16* A = out16 + (size_t)(mt * 16) * DIM;
    const _Float16* B = crt   + (size_t)(nt * 16) * DIM;
    v8f c = {};
#pragma unroll
    for (int kk = 0; kk < DIM; kk += 32)
        c = wmma16(load_frag(A + kk, DIM, lane), load_frag(B + kk, DIM, lane), c);
    int col   = nt * 16 + (lane & 15);
    float bv  = cbias[col];
    int rbase = mt * 16 + ((lane < 16) ? 0 : 8);
#pragma unroll
    for (int v = 0; v < 8; ++v) {
        int row = rbase + v;
        float dn  = fmaxf(cnt[row], 1.0f);
        float val = lrelu(agg[(size_t)row * DIM + col] / dn + c[v] + bv);
        m16[(size_t)row * DIM + col] = (_Float16)val;
    }
}

// gi = m @ W_ih^T + b_ih ; gh = h @ W_hh^T + b_hh   (both [N,192], WMMA K=64)
__global__ void k_gru_mm(const _Float16* __restrict__ m16, const _Float16* __restrict__ out16,
                         const _Float16* __restrict__ wih, const _Float16* __restrict__ whh,
                         const float* __restrict__ bih, const float* __restrict__ bhh,
                         float* __restrict__ gi, float* __restrict__ gh, int N) {
    int lane = threadIdx.x & 31;
    int tile = blockIdx.x * 8 + (threadIdx.x >> 5);
    int mtiles = N >> 4;
    if (tile >= mtiles * 24) return;
    int which = tile / (mtiles * 12);            // 0 -> gi, 1 -> gh (wave-uniform)
    int t2 = tile - which * mtiles * 12;
    int mt = t2 / 12, nt = t2 % 12;
    const _Float16* A = (which == 0 ? m16 : out16) + (size_t)(mt * 16) * DIM;
    const _Float16* B = (which == 0 ? wih : whh)   + (size_t)(nt * 16) * DIM;
    const float* bias = which == 0 ? bih : bhh;
    float* G = which == 0 ? gi : gh;
    v8f c = {};
#pragma unroll
    for (int kk = 0; kk < DIM; kk += 32)
        c = wmma16(load_frag(A + kk, DIM, lane), load_frag(B + kk, DIM, lane), c);
    int col   = nt * 16 + (lane & 15);
    float bv  = bias[col];
    int rbase = mt * 16 + ((lane < 16) ? 0 : 8);
#pragma unroll
    for (int v = 0; v < 8; ++v)
        G[(size_t)(rbase + v) * 192 + col] = c[v] + bv;
}

// GRU elementwise update; refresh h (f32) and out16 (f16 copy for next WMMA A).
__global__ void k_gru_elem(const float* __restrict__ gi, const float* __restrict__ gh,
                           float* __restrict__ h, _Float16* __restrict__ out16, int N) {
    int idx = blockIdx.x * blockDim.x + threadIdx.x;
    if (idx >= N * DIM) return;
    int n = idx >> 6, d = idx & 63;
    const float* gip = gi + (size_t)n * 192;
    const float* ghp = gh + (size_t)n * 192;
    float r    = sigm(gip[d] + ghp[d]);
    float z    = sigm(gip[64 + d] + ghp[64 + d]);
    float cand = tanhf(gip[128 + d] + r * ghp[128 + d]);
    float hn = (1.f - z) * cand + z * h[idx];
    h[idx] = hn;
    out16[idx] = (_Float16)hn;
}

// Set2Set LSTM cell: one block per graph, 256 threads = 256 gates (i,f,g,o).
__global__ void k_s2s_cell(const float* __restrict__ qstar, float* __restrict__ hs,
                           float* __restrict__ cs,
                           const float* __restrict__ wih, const float* __restrict__ whh,
                           const float* __restrict__ bih, const float* __restrict__ bhh) {
    int b = blockIdx.x, t = threadIdx.x;
    __shared__ float gates[256];
    float acc = bih[t] + bhh[t];
    const float* q  = qstar + b * 128;
    const float* hb = hs + b * 64;
#pragma unroll 4
    for (int j = 0; j < 128; ++j) acc += q[j] * wih[t * 128 + j];
#pragma unroll 4
    for (int j = 0; j < 64;  ++j) acc += hb[j] * whh[t * 64 + j];
    gates[t] = acc;
    __syncthreads();
    if (t < 64) {
        float i_ = gates[t], f_ = gates[64 + t], g_ = gates[128 + t], o_ = gates[192 + t];
        float c = sigm(f_) * cs[b * 64 + t] + sigm(i_) * tanhf(g_);
        cs[b * 64 + t] = c;
        hs[b * 64 + t] = sigm(o_) * tanhf(c);
    }
}

// Segment softmax attention + readout; graphs are contiguous runs of npg nodes.
__global__ void k_s2s_attn(const float* __restrict__ h, const float* __restrict__ hs,
                           float* __restrict__ qstar, int npg) {
    int b = blockIdx.x, t = threadIdx.x;        // 64 threads (dim index)
    __shared__ float sh[64];
    __shared__ float ea[256];                   // npg <= 256
    __shared__ float stat[2];
    sh[t] = hs[b * 64 + t];
    __syncthreads();
    int n0 = b * npg;
    for (int n = t; n < npg; n += 64) {
        const float* hp = h + (size_t)(n0 + n) * 64;
        float acc = 0.f;
#pragma unroll 4
        for (int d = 0; d < 64; ++d) acc += hp[d] * sh[d];
        ea[n] = acc;
    }
    __syncthreads();
    if (t == 0) {
        float mx = -1e30f;
        for (int n = 0; n < npg; ++n) mx = fmaxf(mx, ea[n]);
        stat[0] = mx;
    }
    __syncthreads();
    for (int n = t; n < npg; n += 64) ea[n] = expf(ea[n] - stat[0]);
    __syncthreads();
    if (t == 0) {
        float s = 0.f;
        for (int n = 0; n < npg; ++n) s += ea[n];
        stat[1] = s;
    }
    __syncthreads();
    float inv = 1.f / stat[1];
    float acc = 0.f;
    for (int n = 0; n < npg; ++n) acc += ea[n] * h[(size_t)(n0 + n) * 64 + t];
    qstar[b * 128 + t]      = sh[t];
    qstar[b * 128 + 64 + t] = acc * inv;
}

__global__ void k_final(const float* __restrict__ qstar, const float* __restrict__ w,
                        const float* __restrict__ bias, float* __restrict__ out, int B) {
    int b = blockIdx.x * blockDim.x + threadIdx.x;
    if (b >= B) return;
    float acc = bias[0];
    const float* q = qstar + b * 128;
#pragma unroll 4
    for (int j = 0; j < 128; ++j) acc += q[j] * w[j];
    out[b] = acc;
}

// ------------------------------- launcher ----------------------------------

extern "C" void kernel_launch(void* const* d_in, const int* in_sizes, int n_in,
                              void* d_out, int out_size, void* d_ws, size_t ws_size,
                              hipStream_t stream) {
    const float* x        = (const float*)d_in[0];
    const float* ea       = (const float*)d_in[1];
    const float* lin0_w   = (const float*)d_in[2];
    const float* lin0_b   = (const float*)d_in[3];
    const float* enet_w1  = (const float*)d_in[4];
    const float* enet_b1  = (const float*)d_in[5];
    const float* enet_w2  = (const float*)d_in[6];
    const float* enet_b2  = (const float*)d_in[7];
    const float* conv_root= (const float*)d_in[8];
    const float* conv_bias= (const float*)d_in[9];
    const float* gru_w_ih = (const float*)d_in[10];
    const float* gru_w_hh = (const float*)d_in[11];
    const float* gru_b_ih = (const float*)d_in[12];
    const float* gru_b_hh = (const float*)d_in[13];
    const float* s2s_w_ih = (const float*)d_in[14];
    const float* s2s_w_hh = (const float*)d_in[15];
    const float* s2s_b_ih = (const float*)d_in[16];
    const float* s2s_b_hh = (const float*)d_in[17];
    const float* lin1_w   = (const float*)d_in[18];
    const float* lin1_b   = (const float*)d_in[19];
    const int*   eidx     = (const int*)d_in[20];
    (void)n_in; (void)ws_size;

    const int N   = in_sizes[0] / NF;      // 10000
    const int E   = in_sizes[1] / 4;       // 20000
    const int B   = out_size;              // n_graphs (scalar lives on device; out is [B,1])
    const int npg = N / B;                 // contiguous nodes per graph
    const int* src = eidx;
    const int* dst = eidx + E;

    // workspace carve (256B aligned)
    char* p = (char*)d_ws;
    auto carve = [&](size_t bytes) { void* r = p; p += (bytes + 255) & ~(size_t)255; return r; };
    float*    h     = (float*)   carve((size_t)N * DIM * 4);
    _Float16* out16 = (_Float16*)carve((size_t)N * DIM * 2);
    _Float16* m16   = (_Float16*)carve((size_t)N * DIM * 2);
    float*    agg   = (float*)   carve((size_t)N * DIM * 4);
    float*    gi    = (float*)   carve((size_t)N * 192 * 4);
    float*    gh    = (float*)   carve((size_t)N * 192 * 4);
    float*    cnt   = (float*)   carve((size_t)N * 4);
    _Float16* g16   = (_Float16*)carve((size_t)E * 128 * 2);
    _Float16* w2t   = (_Float16*)carve((size_t)4096 * 128 * 2);
    _Float16* crt   = (_Float16*)carve((size_t)64 * 64 * 2);
    _Float16* wih16 = (_Float16*)carve((size_t)192 * DIM * 2);
    _Float16* whh16 = (_Float16*)carve((size_t)192 * DIM * 2);
    float*    hs    = (float*)   carve((size_t)B * 64 * 4);
    float*    cs    = (float*)   carve((size_t)B * 64 * 4);
    float*    qstar = (float*)   carve((size_t)B * 128 * 4);
    _Float16* We    = (_Float16*)carve((size_t)E * 4096 * 2);   // 164 MB, L2-resident

    const int TB = 256;
    auto nb = [](int n, int b) { return (n + b - 1) / b; };

    // degree counts (scatter-mean denominator)
    k_zero<<<nb(N, TB), TB, 0, stream>>>(cnt, N);
    k_count<<<nb(E, TB), TB, 0, stream>>>(dst, cnt, E);

    // weight preps (f16, K-contiguous Bt layouts)
    k_transpose_cast16<<<nb(4096 * 128, TB), TB, 0, stream>>>(w2t, enet_w2, 4096, 128);
    k_transpose_cast16<<<nb(64 * 64, TB), TB, 0, stream>>>(crt, conv_root, 64, 64);
    k_cast16<<<nb(192 * DIM, TB), TB, 0, stream>>>(wih16, gru_w_ih, 192 * DIM);
    k_cast16<<<nb(192 * DIM, TB), TB, 0, stream>>>(whh16, gru_w_hh, 192 * DIM);

    // lin0 + edge network
    k_lin0<<<nb(N * DIM, TB), TB, 0, stream>>>(x, lin0_w, lin0_b, h, out16, N);
    k_enet1<<<nb(E * 128, TB), TB, 0, stream>>>(ea, enet_w1, enet_b1, g16, E);
    k_enet2<<<nb((E >> 4) * 64, 8), 256, 0, stream>>>(g16, w2t, enet_b2, We, E);

    // 4 message-passing + GRU steps
    const int mtiles = N >> 4;
    for (int s = 0; s < 4; ++s) {
        k_zero<<<nb(N * DIM, TB), TB, 0, stream>>>(agg, N * DIM);
        k_msg<<<nb(E * 8, TB), TB, 0, stream>>>(h, We, src, dst, agg, E);
        k_conv_m<<<nb(mtiles * 4, 8), 256, 0, stream>>>(out16, crt, agg, cnt, conv_bias, m16, N);
        k_gru_mm<<<nb(mtiles * 24, 8), 256, 0, stream>>>(m16, out16, wih16, whh16,
                                                         gru_b_ih, gru_b_hh, gi, gh, N);
        k_gru_elem<<<nb(N * DIM, TB), TB, 0, stream>>>(gi, gh, h, out16, N);
    }

    // Set2Set pooling
    k_zero<<<nb(B * 64, TB), TB, 0, stream>>>(hs, B * 64);
    k_zero<<<nb(B * 64, TB), TB, 0, stream>>>(cs, B * 64);
    k_zero<<<nb(B * 128, TB), TB, 0, stream>>>(qstar, B * 128);
    for (int s = 0; s < 3; ++s) {
        k_s2s_cell<<<B, 256, 0, stream>>>(qstar, hs, cs, s2s_w_ih, s2s_w_hh, s2s_b_ih, s2s_b_hh);
        k_s2s_attn<<<B, 64, 0, stream>>>(h, hs, qstar, npg);
    }
    k_final<<<nb(B, TB), TB, 0, stream>>>(qstar, lin1_w, lin1_b, (float*)d_out, B);
}
